// VNNConvD_51170240364923
// MI455X (gfx1250) — matrix operand order repaired and checked by reading, not compiled, not run on hardware
//
#include <hip/hip_runtime.h>

#define BB      8
#define NN      8192
#define NPOINT  2048
#define NSAMPLE 16
#define CIN     22
#define COUT    64
#define EPSV    1e-6f
#define NEGS    0.2f

#define OUT_NEWXYZ 0
#define OUT_FEATS  (BB * 3 * NPOINT)                       // 49152
#define OUT_FPS    (OUT_FEATS + BB * 3 * COUT * NPOINT)    // 3194880

typedef __attribute__((ext_vector_type(16))) _Float16 v16h;
typedef __attribute__((ext_vector_type(8)))  _Float16 v8h;
typedef __attribute__((ext_vector_type(8)))  float    v8f;

// ---------------------------------------------------------------------------
// Kernel 1: furthest point sampling. One block per batch; dmin lives in
// registers (8 points per thread), block-wide argmax via LDS tree each step.
// Writes fps_idx (as float) and new_xyz ([B,3,S]) directly to d_out.
// ---------------------------------------------------------------------------
__global__ __launch_bounds__(1024) void fps_kernel(const float* __restrict__ xyz,
                                                   float* __restrict__ out) {
  const int b   = blockIdx.x;
  const int tid = threadIdx.x;
  const float* xb = xyz + (size_t)b * 3 * NN;

  float px[8], py[8], pz[8], dmin[8];
#pragma unroll
  for (int r = 0; r < 8; ++r) {
    const int i = tid * 8 + r;
    px[r] = xb[i]; py[r] = xb[NN + i]; pz[r] = xb[2 * NN + i];
    dmin[r] = 1e10f;
  }

  __shared__ float sval[1024];
  __shared__ int   sidx[1024];
  __shared__ float cent[3];

  float* onx  = out + OUT_NEWXYZ + (size_t)b * 3 * NPOINT;
  float* ofps = out + OUT_FPS    + (size_t)b * NPOINT;

  int far = 0;
  for (int it = 0; it < NPOINT; ++it) {
    if (tid == 0) {
      ofps[it] = (float)far;
      const float cx = xb[far], cy = xb[NN + far], cz = xb[2 * NN + far];
      cent[0] = cx; cent[1] = cy; cent[2] = cz;
      onx[0 * NPOINT + it] = cx;
      onx[1 * NPOINT + it] = cy;
      onx[2 * NPOINT + it] = cz;
    }
    __syncthreads();
    const float cx = cent[0], cy = cent[1], cz = cent[2];

    float best = -1.0f; int bi = tid * 8;
#pragma unroll
    for (int r = 0; r < 8; ++r) {
      const float dx = px[r] - cx, dy = py[r] - cy, dz = pz[r] - cz;
      const float d = dx * dx + dy * dy + dz * dz;
      dmin[r] = fminf(dmin[r], d);
      if (dmin[r] > best) { best = dmin[r]; bi = tid * 8 + r; }
    }
    sval[tid] = best; sidx[tid] = bi;
    __syncthreads();
    for (int s = 512; s > 0; s >>= 1) {
      if (tid < s) {
        const float v2 = sval[tid + s]; const int i2 = sidx[tid + s];
        if (v2 > sval[tid] || (v2 == sval[tid] && i2 < sidx[tid])) {
          sval[tid] = v2; sidx[tid] = i2;
        }
      }
      __syncthreads();
    }
    far = sidx[0];
    __syncthreads();
  }
}

// ---------------------------------------------------------------------------
// Kernel 2: 16-NN per sampled point. One block per (b,s); each thread keeps a
// sorted local top-16 over a strided slice of N, then 16 rounds of block-wide
// argmin selection in LDS. Order within the 16 is irrelevant (max-pool later).
// ---------------------------------------------------------------------------
__global__ __launch_bounds__(256) void knn_kernel(const float* __restrict__ xyz,
                                                  const float* __restrict__ out,
                                                  int* __restrict__ knn_ws) {
  const int bs = blockIdx.x;
  const int b  = bs / NPOINT;
  const int s  = bs % NPOINT;
  const int tid = threadIdx.x;
  const float* xb = xyz + (size_t)b * 3 * NN;

  const float qx = out[OUT_NEWXYZ + (size_t)b * 3 * NPOINT + 0 * NPOINT + s];
  const float qy = out[OUT_NEWXYZ + (size_t)b * 3 * NPOINT + 1 * NPOINT + s];
  const float qz = out[OUT_NEWXYZ + (size_t)b * 3 * NPOINT + 2 * NPOINT + s];

  float bd[16]; int bi[16];
#pragma unroll
  for (int j = 0; j < 16; ++j) { bd[j] = 3.4e38f; bi[j] = 0; }

  for (int r = 0; r < NN / 256; ++r) {
    const int i = tid + r * 256;
    const float dx = xb[i] - qx, dy = xb[NN + i] - qy, dz = xb[2 * NN + i] - qz;
    const float d = dx * dx + dy * dy + dz * dz;
    if (d < bd[15]) {
      int j = 15;
      while (j > 0 && bd[j - 1] > d) { bd[j] = bd[j - 1]; bi[j] = bi[j - 1]; --j; }
      bd[j] = d; bi[j] = i;
    }
  }

  __shared__ float sd[256 * 16];
  __shared__ int   si[256 * 16];
  __shared__ float rv[256];
  __shared__ int   rp[256];
#pragma unroll
  for (int j = 0; j < 16; ++j) { sd[tid * 16 + j] = bd[j]; si[tid * 16 + j] = bi[j]; }
  __syncthreads();

  for (int round = 0; round < NSAMPLE; ++round) {
    float mv = 3.4e38f; int mp = tid * 16;
#pragma unroll
    for (int j = 0; j < 16; ++j) {
      const float v = sd[tid * 16 + j];
      if (v < mv) { mv = v; mp = tid * 16 + j; }
    }
    rv[tid] = mv; rp[tid] = mp;
    __syncthreads();
    for (int st = 128; st > 0; st >>= 1) {
      if (tid < st) {
        if (rv[tid + st] < rv[tid] ||
            (rv[tid + st] == rv[tid] && rp[tid + st] < rp[tid])) {
          rv[tid] = rv[tid + st]; rp[tid] = rp[tid + st];
        }
      }
      __syncthreads();
    }
    const int wpos = rp[0];
    if (tid == 0) knn_ws[(size_t)bs * NSAMPLE + round] = si[wpos];
    if ((wpos >> 4) == tid) sd[wpos] = 3.4e38f;
    __syncthreads();
  }
}

// ---------------------------------------------------------------------------
// Kernel 3: fused gather + VN linear-leakyReLU + VN max-pool, one block per
// (b,s). 128 threads = 4 waves; wave w owns output rows [16w,16w+16).
// P = w_feat@X, D = w_dir@X via v_wmma_f32_16x16x32_f16 (K: 22 -> pad 32),
// dot/dsq over v are lane-local, pool GEMM chains two K=32 WMMAs (K=64).
// Projection coefficient uses v_rcp_f32 (branchless) instead of IEEE divide.
// ---------------------------------------------------------------------------
__global__ __launch_bounds__(128) void vn_kernel(const float* __restrict__ xyz,
                                                 const float* __restrict__ pts,
                                                 const float* __restrict__ w_feat,
                                                 const float* __restrict__ w_dir,
                                                 const float* __restrict__ w_pool,
                                                 const int* __restrict__ knn_ws,
                                                 float* __restrict__ out) {
  const int bs   = blockIdx.x;
  const int b    = bs / NPOINT;
  const int s    = bs % NPOINT;
  const int tid  = threadIdx.x;
  const int lane = tid & 31;
  const int wv   = tid >> 5;

  __shared__ __align__(16) _Float16 wfs[COUT][32];       // w_feat, K padded
  __shared__ __align__(16) _Float16 wds[COUT][32];       // w_dir,  K padded
  __shared__ __align__(16) _Float16 wps[COUT][64];       // w_pool
  __shared__ __align__(16) _Float16 xbm[3][16][32];      // B-mat: [v][col=k][K=c]
  __shared__ __align__(16) _Float16 xob[3][16][64];      // pool B-mat: [v][col=k][K=o]
  __shared__ float xof[COUT][3][16];                     // x_out f32 for pooled gather
  __shared__ float scs[COUT][16];                        // scores
  __shared__ int   nix[16];
  __shared__ float qs[3];

  for (int t = tid; t < COUT * 32; t += 128) {
    const int o = t >> 5, k = t & 31;
    wfs[o][k] = (_Float16)(k < CIN ? w_feat[o * CIN + k] : 0.0f);
    wds[o][k] = (_Float16)(k < CIN ? w_dir [o * CIN + k] : 0.0f);
  }
  for (int t = tid; t < COUT * 64; t += 128) {
    wps[t >> 6][t & 63] = (_Float16)w_pool[t];
  }
  if (tid < NSAMPLE) nix[tid] = knn_ws[(size_t)bs * NSAMPLE + tid];
  if (tid < 3)       qs[tid]  = out[OUT_NEWXYZ + (size_t)b * 3 * NPOINT + tid * NPOINT + s];
  __syncthreads();

  // Gather X: channel c of vector-component v at neighbor k.
  const float* xb = xyz + (size_t)b * 3 * NN;
  const float* pb = pts + (size_t)b * 3 * (CIN - 1) * NN;
  for (int t = tid; t < 3 * 16 * 32; t += 128) {
    const int v = t / (16 * 32);
    const int k = (t >> 5) & 15;
    const int c = t & 31;
    float val = 0.0f;
    const int n = nix[k];
    if (c == 0)        val = xb[v * NN + n] - qs[v];
    else if (c < CIN)  val = pb[(3 * (c - 1) + v) * NN + n];
    xbm[v][k][c] = (_Float16)val;
  }
  __syncthreads();

  // Fragment addressing (16-bit WMMA layouts, wave32).
  const int row   = wv * 16 + (lane & 15);   // A-matrix row for this lane
  const int abase = (lane >= 16) ? 8 : 0;    // A K-subgroup base
  const int bhalf = (lane >= 16) ? 16 : 0;   // unused; B loads use abase
  const int col   = lane & 15;               // B/C/D column
  const int rb    = wv * 16 + ((lane >= 16) ? 8 : 0);  // C/D row base

  v16h afeat, adir;
  *((v8h*)&afeat)     = *(const v8h*)&wfs[row][abase];
  *((v8h*)&afeat + 1) = *(const v8h*)&wfs[row][16 + abase];
  *((v8h*)&adir)      = *(const v8h*)&wds[row][abase];
  *((v8h*)&adir + 1)  = *(const v8h*)&wds[row][16 + abase];

  v8f pf[3], df[3];
#pragma unroll
  for (int v = 0; v < 3; ++v) {
    v16h bx;
    *((v8h*)&bx)     = *(const v8h*)&xbm[v][col][abase];
    *((v8h*)&bx + 1) = *(const v8h*)&xbm[v][col][16 + abase];
    v8f z0 = {};
    pf[v] = __builtin_amdgcn_wmma_f32_16x16x32_f16(false, afeat, false, bx,
                                                   (short)0, z0, false, false);
    v8f z1 = {};
    df[v] = __builtin_amdgcn_wmma_f32_16x16x32_f16(false, adir, false, bx,
                                                   (short)0, z1, false, false);
  }

  // VN leaky-ReLU: reductions over v are lane-local across the 3 accumulators.
  // coef = (dot >= 0) ? 0 : dot/dsq  ==  min(dot, 0) * rcp(dsq)   (dsq >= eps)
  float xo[3][8];
#pragma unroll
  for (int j = 0; j < 8; ++j) {
    const float dot = pf[0][j] * df[0][j] + pf[1][j] * df[1][j] + pf[2][j] * df[2][j];
    const float dsq = df[0][j] * df[0][j] + df[1][j] * df[1][j] + df[2][j] * df[2][j] + EPSV;
    const float coef = fminf(dot, 0.0f) * __builtin_amdgcn_rcpf(dsq);
#pragma unroll
    for (int v = 0; v < 3; ++v) {
      const float p = pf[v][j], d = df[v][j];
      xo[v][j] = NEGS * p + (1.0f - NEGS) * (p - coef * d);
    }
  }

  // Stage x_out: f16 B-matrix layout for the pool GEMM + f32 copy for output.
#pragma unroll
  for (int v = 0; v < 3; ++v) {
    v8h hv;
#pragma unroll
    for (int j = 0; j < 8; ++j) hv[j] = (_Float16)xo[v][j];
    *(v8h*)&xob[v][col][rb] = hv;
#pragma unroll
    for (int j = 0; j < 8; ++j) xof[rb + j][v][col] = xo[v][j];
  }
  __syncthreads();

  // Pool GEMM: dp = w_pool @ x_out, K = 64 as two chained K=32 WMMAs.
  v16h a0, a1;
  *((v8h*)&a0)     = *(const v8h*)&wps[row][abase];
  *((v8h*)&a0 + 1) = *(const v8h*)&wps[row][16 + abase];
  *((v8h*)&a1)     = *(const v8h*)&wps[row][32 + abase];
  *((v8h*)&a1 + 1) = *(const v8h*)&wps[row][48 + abase];

  v8f dp[3];
#pragma unroll
  for (int v = 0; v < 3; ++v) {
    v16h b0, b1;
    *((v8h*)&b0)     = *(const v8h*)&xob[v][col][abase];
    *((v8h*)&b0 + 1) = *(const v8h*)&xob[v][col][16 + abase];
    *((v8h*)&b1)     = *(const v8h*)&xob[v][col][32 + abase];
    *((v8h*)&b1 + 1) = *(const v8h*)&xob[v][col][48 + abase];
    v8f c = {};
    c = __builtin_amdgcn_wmma_f32_16x16x32_f16(false, a0, false, b0, (short)0, c, false, false);
    c = __builtin_amdgcn_wmma_f32_16x16x32_f16(false, a1, false, b1, (short)0, c, false, false);
    dp[v] = c;
  }

  // score[o][k] = sum_v x_out * dp (lane-local), then argmax over k + gather.
#pragma unroll
  for (int j = 0; j < 8; ++j) {
    scs[rb + j][col] = xo[0][j] * dp[0][j] + xo[1][j] * dp[1][j] + xo[2][j] * dp[2][j];
  }
  __syncthreads();

  if (tid < COUT) {
    const int o = tid;
    float bv = scs[o][0]; int bk = 0;
#pragma unroll
    for (int k = 1; k < NSAMPLE; ++k) {
      if (scs[o][k] > bv) { bv = scs[o][k]; bk = k; }
    }
#pragma unroll
    for (int v = 0; v < 3; ++v) {
      out[OUT_FEATS + ((size_t)b * 3 * COUT + o * 3 + v) * NPOINT + s] = xof[o][v][bk];
    }
  }
}

// ---------------------------------------------------------------------------
extern "C" void kernel_launch(void* const* d_in, const int* in_sizes, int n_in,
                              void* d_out, int out_size, void* d_ws, size_t ws_size,
                              hipStream_t stream) {
  (void)in_sizes; (void)n_in; (void)out_size; (void)ws_size;
  const float* xyz    = (const float*)d_in[0];
  const float* points = (const float*)d_in[1];
  const float* w_feat = (const float*)d_in[2];
  const float* w_dir  = (const float*)d_in[3];
  const float* w_pool = (const float*)d_in[4];
  float* out = (float*)d_out;
  int* knn_ws = (int*)d_ws;   // B*NPOINT*NSAMPLE ints = 1 MB

  fps_kernel<<<BB, 1024, 0, stream>>>(xyz, out);
  knn_kernel<<<BB * NPOINT, 256, 0, stream>>>(xyz, out, knn_ws);
  vn_kernel<<<BB * NPOINT, 128, 0, stream>>>(xyz, points, w_feat, w_dir, w_pool,
                                             knn_ws, out);
}
